// GCN_DAE_85109071938338
// MI455X (gfx1250) — matrix operand order repaired and checked by reading, not compiled, not run on hardware
//
#include <hip/hip_runtime.h>
#include <hip/hip_bf16.h>

// ---------------------------------------------------------------------------
// Types for CDNA5 WMMA (wave32, v_wmma_f32_16x16x32_bf16) + TDM descriptors
// ---------------------------------------------------------------------------
typedef __attribute__((ext_vector_type(16))) __bf16 v16bf;
typedef __attribute__((ext_vector_type(2)))  __bf16 v2bf;
typedef __attribute__((ext_vector_type(8)))  float  v8f;
typedef __attribute__((ext_vector_type(4)))  unsigned u32x4;
typedef __attribute__((ext_vector_type(8)))  int      i32x8;
typedef __attribute__((ext_vector_type(4)))  int      i32x4;

#define TK 30

#if defined(__HIP_DEVICE_COMPILE__) && __has_builtin(__builtin_amdgcn_tensor_load_to_lds) && \
    __has_builtin(__builtin_amdgcn_s_wait_tensorcnt)
#define HAS_TDM 1
#else
#define HAS_TDM 0
#endif

// ---------------------------------------------------------------------------
// TDM: DMA a 2D bf16 tile (tileRows x rowElems, row-major, contiguous) from
// global memory into LDS, with optional LDS row padding (D# pad fields).
// One tensor_load_to_lds per call; caller gates to one wave.
// ---------------------------------------------------------------------------
__device__ __forceinline__ void tdm_load_tile_2d_bf16(const void* gsrc, void* ldsDst,
                                                      unsigned tileRows, unsigned rowElems,
                                                      bool padEn, unsigned padIntCode,
                                                      unsigned padAmtCode) {
#if HAS_TDM
    const unsigned long long g = (unsigned long long)gsrc;
    const unsigned lds = (unsigned)(unsigned long long)ldsDst;   // LDS offset = addr[31:0]

    u32x4 g0;
    g0[0] = 1u;                                                  // count=1 (valid user D#)
    g0[1] = lds;                                                 // lds_addr
    g0[2] = (unsigned)g;                                         // global_addr[31:0]
    g0[3] = (unsigned)((g >> 32) & 0x01FFFFFFu) | (2u << 30);    // global_addr[56:32] | type=2

    i32x8 g1;
    unsigned d0 = (1u << 16);                                    // data_size = 1 -> 2 bytes
    if (padEn) d0 |= (1u << 20) | (padIntCode << 22) | (padAmtCode << 25);
    g1[0] = (int)d0;                                             // wg_mask=0 (no cluster)
    g1[1] = (int)((rowElems & 0xFFFFu) << 16);                   // tensor_dim0 lo16 @bits63:48
    g1[2] = (int)(((rowElems >> 16) & 0xFFFFu) |                 // tensor_dim0 hi16
                  ((tileRows & 0xFFFFu) << 16));                 // tensor_dim1 lo16
    g1[3] = (int)(((tileRows >> 16) & 0xFFFFu) |                 // tensor_dim1 hi16
                  ((rowElems & 0xFFFFu) << 16));                 // tile_dim0
    g1[4] = (int)(tileRows & 0xFFFFu);                           // tile_dim1 (tile_dim2=0 -> 2D)
    g1[5] = (int)rowElems;                                       // tensor_dim0_stride lo32
    g1[6] = 0;                                                   // stride hi / dim1_stride (unused, 2D)
    g1[7] = 0;

    const i32x4 z4 = {0, 0, 0, 0};
#if __clang_major__ >= 23
    const i32x8 z8 = {0, 0, 0, 0, 0, 0, 0, 0};
    __builtin_amdgcn_tensor_load_to_lds(g0, g1, z4, z4, z8, 0);
#else
    __builtin_amdgcn_tensor_load_to_lds(g0, g1, z4, z4, 0);
#endif
    __builtin_amdgcn_s_wait_tensorcnt(0);
#else
    (void)gsrc; (void)ldsDst; (void)tileRows; (void)rowElems;
    (void)padEn; (void)padIntCode; (void)padAmtCode;
#endif
}

// ---------------------------------------------------------------------------
// Generic GEMM: C[M x NT*16] = act( A[M x Kd] (f32) @ B[Kd x NT*16] (f32) + bias )
// B panel cooperatively staged (f32 -> bf16) into padded LDS each k-step and
// shared by all 8 waves; inner loop = 1 aligned v16bf LDS read per tile.
// A converted f32->bf16 in flight (it is the streaming 400MB operand).
// ---------------------------------------------------------------------------
template<int NT, bool RELU, bool BIAS>
__global__ void gemm_wmma(const float* __restrict__ A, long lda,
                          const float* __restrict__ B, long ldb,
                          const float* __restrict__ bias,
                          float* __restrict__ C, long ldc,
                          int M, int Kd) {
    constexpr int NN = NT * 16;
    constexpr int BST = 48;                       // padded k-stride (breaks bank conflicts, 32B-aligned)
    __shared__ __bf16 Bs[NN * BST];

    const int tid  = threadIdx.x;
    const int lane = tid & 31;
    const int wave = tid >> 5;
    const int strip = blockIdx.x * 8 + wave;      // wave-uniform 16-row strip
    const int row0  = strip * 16;
    const bool active = (row0 < M);
    const int n    = lane & 15;
    const int half = lane >> 4;

    v8f acc[NT];
#pragma unroll
    for (int t = 0; t < NT; ++t) {
        const float bv = BIAS ? bias[t * 16 + n] : 0.0f;
#pragma unroll
        for (int e = 0; e < 8; ++e) acc[t][e] = bv;
    }

    const float* Arow = A + (long)(row0 + n) * lda;

    for (int k0 = 0; k0 < Kd; k0 += 32) {
        __syncthreads();                           // previous-iteration reads done
        // ---- cooperative stage: B[k0..k0+31][0..NN) -> bf16 LDS (transposed) ----
        for (int e = tid; e < 32 * NN; e += 256) {
            const int k   = e / NN;
            const int col = e % NN;
            const float v = (k0 + k < Kd) ? B[(long)(k0 + k) * ldb + col] : 0.0f;
            Bs[col * BST + k] = (__bf16)v;
        }
        __syncthreads();

        if (active) {
            // ---- A fragment: 16x32 bf16 (lane m = n; K pair per VGPR r) ----
            v16bf a;
#pragma unroll
            for (int r = 0; r < 8; ++r) {
                const int kb = r * 2 + half * 16;
                float x0 = 0.0f, x1 = 0.0f;
                if (k0 + kb < Kd) {                // pair-aligned tail mask (Kd even)
                    const float2 v = *(const float2*)(Arow + k0 + kb);
                    x0 = v.x; x1 = v.y;
                }
                a[2 * r]     = (__bf16)x0;
                a[2 * r + 1] = (__bf16)x1;
            }
#pragma unroll
            for (int t = 0; t < NT; ++t) {
                const v16bf b = *(const v16bf*)&Bs[(t * 16 + n) * BST + half * 16];
                acc[t] = __builtin_amdgcn_wmma_f32_16x16x32_bf16(
                    false, a, false, b, (short)0, acc[t], false, false);
            }
        }
    }

    if (active) {
#pragma unroll
        for (int t = 0; t < NT; ++t) {
#pragma unroll
            for (int r = 0; r < 8; ++r) {
                float v = acc[t][r];
                if (RELU) v = v > 0.0f ? v : 0.0f;
                C[(long)(row0 + r + half * 8) * ldc + (t * 16 + n)] = v;
            }
        }
    }
}

// ---------------------------------------------------------------------------
// Row L2-normalize h (f32) -> hn (bf16). One wave per row, D = 256.
// ---------------------------------------------------------------------------
__global__ void rownorm_kernel(const float* __restrict__ h,
                               __bf16* __restrict__ hn, int N, int D) {
    const int lane = threadIdx.x & 31;
    const int row  = blockIdx.x * (blockDim.x >> 5) + (threadIdx.x >> 5);
    if (row >= N) return;
    const float* hr = h + (long)row * D;
    float ss = 0.0f;
    for (int i = lane; i < D; i += 32) { const float v = hr[i]; ss += v * v; }
#pragma unroll
    for (int off = 16; off > 0; off >>= 1) ss += __shfl_xor(ss, off, 32);
    const float inv = 1.0f / (sqrtf(ss) + 1e-8f);
    __bf16* o = hn + (long)row * D;
    for (int i = lane; i < D; i += 32) o[i] = (__bf16)(hr[i] * inv);
}

// ---------------------------------------------------------------------------
// S = relu(hn @ hn^T). Block = one 16-col panel (TDM -> LDS, padded rows,
// shared by 8 row-tile waves) x 8 row tiles. K = 256, 8 WMMAs per tile.
// ---------------------------------------------------------------------------
#define SIMP 272   // 256 + 16 bf16 pad per LDS row (TDM pad: 128 DW interval, 8 DW amount)

__global__ void sim_wmma(const __bf16* __restrict__ hn,
                         float* __restrict__ S, int N) {
    __shared__ __bf16 Bs[16 * SIMP];
    const int tid  = threadIdx.x;
    const int lane = tid & 31;
    const int wave = tid >> 5;
    const int ntiles = N / 16;
    const int col0 = blockIdx.x * 16;
    const int rowTile = blockIdx.y * 8 + wave;     // wave-uniform
    const __bf16* src = hn + (long)col0 * 256;

#if HAS_TDM
    if (wave == 0) {
        // pad_interval code 6 = 128 DWORDs (one 256-elem bf16 row),
        // pad_amount  code 7 = 8 DWORDs (16 bf16 elems) -> LDS row stride 272
        tdm_load_tile_2d_bf16(src, Bs, 16, 256, true, 6u, 7u);
    }
#else
    for (int e = tid; e < 16 * 256; e += 256) {
        const int r = e >> 8, k = e & 255;
        Bs[r * SIMP + k] = src[e];
    }
#endif
    __syncthreads();

    if (rowTile < ntiles) {
        const int row0 = rowTile * 16;
        const int n = lane & 15, half = lane >> 4;
        const __bf16* Arow = hn + (long)(row0 + n) * 256;
        v8f acc = {};
        for (int k0 = 0; k0 < 256; k0 += 32) {
            v16bf a;
#pragma unroll
            for (int r = 0; r < 8; ++r) {
                const int kb = r * 2 + half * 16;
                const v2bf pa = *(const v2bf*)(Arow + k0 + kb);
                a[2 * r] = pa[0]; a[2 * r + 1] = pa[1];
            }
            const v16bf b = *(const v16bf*)&Bs[n * SIMP + k0 + half * 16];
            acc = __builtin_amdgcn_wmma_f32_16x16x32_bf16(
                false, a, false, b, (short)0, acc, false, false);
        }
#pragma unroll
        for (int r = 0; r < 8; ++r) {
            float v = acc[r];
            v = v > 0.0f ? v : 0.0f;
            S[(long)(row0 + r + half * 8) * N + (col0 + n)] = v;
        }
    }
}

// ---------------------------------------------------------------------------
// Top-30 per row: row cached in LDS, 30 argmax rounds (tie -> lowest index).
// ---------------------------------------------------------------------------
__global__ void topk_kernel(const float* __restrict__ S,
                            float* __restrict__ vals, int* __restrict__ inds,
                            int N) {
    __shared__ float srow[10016];
    __shared__ float redv[256];
    __shared__ int   redi[256];
    const int row = blockIdx.x;
    const int tid = threadIdx.x;
    const float* sr = S + (long)row * N;
    for (int i = tid; i < N; i += 256) srow[i] = sr[i];
    __syncthreads();
    for (int k = 0; k < TK; ++k) {
        float bv = -1.0f; int bi = -1;               // S >= 0 after relu
        for (int i = tid; i < N; i += 256) {
            const float v = srow[i];
            if (v > bv) { bv = v; bi = i; }
        }
        redv[tid] = bv; redi[tid] = bi;
        __syncthreads();
        for (int s = 128; s > 0; s >>= 1) {
            if (tid < s) {
                const float v2 = redv[tid + s]; const int i2 = redi[tid + s];
                if (v2 > redv[tid] ||
                    (v2 == redv[tid] && (unsigned)i2 < (unsigned)redi[tid])) {
                    redv[tid] = v2; redi[tid] = i2;
                }
            }
            __syncthreads();
        }
        if (tid == 0) {
            const int bidx = redi[0];
            vals[(long)row * TK + k] = redv[0];
            inds[(long)row * TK + k] = bidx;
            srow[bidx] = -3e38f;                     // mask out
        }
        __syncthreads();
    }
}

__global__ void zero_kernel(float4* __restrict__ p, long n4) {
    long i = (long)blockIdx.x * blockDim.x + threadIdx.x;
    const long stride = (long)gridDim.x * blockDim.x;
    const float4 z = {0.0f, 0.0f, 0.0f, 0.0f};
    for (; i < n4; i += stride) p[i] = z;
}

// A = (A_topk + A_topk^T) / 2 via symmetric atomic scatter
__global__ void scatter_kernel(const float* __restrict__ vals,
                               const int* __restrict__ inds,
                               float* __restrict__ adj, int N) {
    const int t = blockIdx.x * 256 + threadIdx.x;
    if (t >= N * TK) return;
    const int r = t / TK;
    const int j = inds[t];
    const float v = 0.5f * vals[t];
    atomicAdd(adj + (long)r * N + j, v);
    atomicAdd(adj + (long)j * N + r, v);
}

__global__ void rowsum_kernel(const float* __restrict__ adj,
                              float* __restrict__ dis, int N) {
    __shared__ float red[256];
    const int row = blockIdx.x, tid = threadIdx.x;
    const float* ar = adj + (long)row * N;
    float s = 0.0f;
    for (int i = tid; i < N; i += 256) s += ar[i];
    red[tid] = s; __syncthreads();
    for (int st = 128; st > 0; st >>= 1) {
        if (tid < st) red[tid] += red[tid + st];
        __syncthreads();
    }
    if (tid == 0) {
        const float d = red[0];
        dis[row] = d > 0.0f ? rsqrtf(fmaxf(d, 1e-8f)) : 0.0f;
    }
}

__global__ void scale_kernel(float* __restrict__ adj,
                             const float* __restrict__ dis, int N) {
    const int col = blockIdx.x * blockDim.x + threadIdx.x;
    const int row = blockIdx.y;
    if (col < N) {
        const long idx = (long)row * N + col;
        adj[idx] *= dis[row] * dis[col];
    }
}

// ---------------------------------------------------------------------------
// Launch
// ---------------------------------------------------------------------------
extern "C" void kernel_launch(void* const* d_in, const int* in_sizes, int n_in,
                              void* d_out, int out_size, void* d_ws, size_t ws_size,
                              hipStream_t stream) {
    (void)in_sizes; (void)n_in; (void)out_size; (void)ws_size;
    const int N = 10000, D = 256, H = 128, C = 16;

    const float* features = (const float*)d_in[0];
    const float* x        = (const float*)d_in[1];
    const float* W1       = (const float*)d_in[2];
    const float* W2       = (const float*)d_in[3];
    const float* Wg1      = (const float*)d_in[4];
    const float* bg1      = (const float*)d_in[5];
    const float* Wg2      = (const float*)d_in[6];
    const float* bg2      = (const float*)d_in[7];

    float* out = (float*)d_out;                 // [N, C]
    float* adj = out + (long)N * C;             // [N, N] (also S scratch)

    char* ws = (char*)d_ws;                     // ~39 MB total
    float*  T1   = (float*)(ws);                         // [N,256] f32
    float*  h    = (float*)(ws + 10240000);              // [N,256] f32
    __bf16* hn   = (__bf16*)(ws + 20480000);             // [N,256] bf16
    float*  XW1  = (float*)(ws + 25600000);              // [N,128] f32
    float*  x1   = (float*)(ws + 30720000);              // [N,128] f32
    float*  XW2  = (float*)(ws + 35840000);              // [N,16]  f32
    float*  vals = (float*)(ws + 36480000);              // [N,30]  f32
    int*    inds = (int*)  (ws + 37680000);              // [N,30]  i32
    float*  dis  = (float*)(ws + 38880000);              // [N]     f32

    const int strips   = N / 16;                         // 625 row strips
    const int gemmGrid = (strips + 7) / 8;               // 8 waves / WG

    // ---- MLP graph learner ----
    gemm_wmma<16, true,  false><<<gemmGrid, 256, 0, stream>>>(features, 256, W1, 256, nullptr, T1, 256, N, 256);
    gemm_wmma<16, false, false><<<gemmGrid, 256, 0, stream>>>(T1,       256, W2, 256, nullptr, h,  256, N, 256);
    rownorm_kernel<<<(N + 7) / 8, 256, 0, stream>>>(h, hn, N, D);

    // ---- similarity + kNN graph ----
    dim3 simGrid(strips, (strips + 7) / 8);              // (625 col panels, 79 row supers)
    sim_wmma<<<simGrid, 256, 0, stream>>>(hn, adj, N);
    topk_kernel<<<N, 256, 0, stream>>>(adj, vals, inds, N);
    zero_kernel<<<4096, 256, 0, stream>>>((float4*)adj, (long)N * N / 4);
    scatter_kernel<<<(N * TK + 255) / 256, 256, 0, stream>>>(vals, inds, adj, N);
    rowsum_kernel<<<N, 256, 0, stream>>>(adj, dis, N);
    dim3 sg((N + 255) / 256, N);
    scale_kernel<<<sg, 256, 0, stream>>>(adj, dis, N);

    // ---- 2-layer GCN ----
    gemm_wmma<8, false, true ><<<gemmGrid, 256, 0, stream>>>(x,   256, Wg1, H, bg1,     XW1, H, N, 256);
    gemm_wmma<8, true,  false><<<gemmGrid, 256, 0, stream>>>(adj, N,   XW1, H, nullptr, x1,  H, N, N);
    gemm_wmma<1, false, true ><<<gemmGrid, 256, 0, stream>>>(x1,  H,   Wg2, C, bg2,     XW2, C, N, H);
    gemm_wmma<1, false, false><<<gemmGrid, 256, 0, stream>>>(adj, N,   XW2, C, nullptr, out, C, N, N);
}
